// MOELayer_16973710753991
// MI455X (gfx1250) — matrix operand order repaired and checked by reading, not compiled, not run on hardware
//
#include <hip/hip_runtime.h>
#include <hip/hip_bf16.h>

#define NUM_EXPERT 16
#define IN_FEAT    1024
#define OUT_FEAT   4096
#define N_TOKENS   2048

#define M_SUPER  64                 // tokens per block iteration (4 WMMA rows tiles)
#define K_CHUNK  128                // K staged in LDS per iteration
#define A_STRIDE (K_CHUNK + 8)      // bf16 elems; 272B rows: 16B-aligned, bank-spread

typedef __attribute__((ext_vector_type(16))) __bf16 v16bf;
typedef __attribute__((ext_vector_type(8)))  __bf16 v8bf;
typedef __attribute__((ext_vector_type(8)))  float  v8f;

union AFrag { v16bf v; v8bf h[2]; };

// Workspace (ints): ws[0..15] counts | ws[16..31] offsets | ws[64..64+N) perm

// ---------------------------------------------------------------------------
// Setup: histogram gates, exclusive scan, scatter token ids grouped by expert.
// ---------------------------------------------------------------------------
__global__ void __launch_bounds__(256)
moe_setup_kernel(const int* __restrict__ gate, int* __restrict__ ws) {
  __shared__ int scnt[NUM_EXPERT];
  __shared__ int soff[NUM_EXPERT];
  const int tid = threadIdx.x;
  if (tid < NUM_EXPERT) scnt[tid] = 0;
  __syncthreads();
  for (int i = tid; i < N_TOKENS; i += 256) atomicAdd(&scnt[gate[i]], 1);
  __syncthreads();
  if (tid == 0) {
    int run = 0;
    for (int e = 0; e < NUM_EXPERT; ++e) {
      ws[e]      = scnt[e];
      ws[16 + e] = run;
      soff[e]    = run;
      run += scnt[e];
      scnt[e] = 0;                      // reuse as scatter cursor
    }
  }
  __syncthreads();
  for (int i = tid; i < N_TOKENS; i += 256) {
    int g = gate[i];
    int pos = soff[g] + atomicAdd(&scnt[g], 1);
    ws[64 + pos] = i;
  }
}

// Load 16 f32 (two float4 pairs at p0,p1) -> bf16 hi + residual lo.
__device__ __forceinline__ void load_cvt16(const float* __restrict__ p0,
                                           const float* __restrict__ p1,
                                           v16bf& hi, v16bf& lo) {
  float x[16];
  *(float4*)(&x[0])  = *(const float4*)(p0);
  *(float4*)(&x[4])  = *(const float4*)(p0 + 4);
  *(float4*)(&x[8])  = *(const float4*)(p1);
  *(float4*)(&x[12]) = *(const float4*)(p1 + 4);
#pragma unroll
  for (int i = 0; i < 16; ++i) {
    __bf16 h = (__bf16)x[i];
    hi[i] = h;
    lo[i] = (__bf16)(x[i] - (float)h);
  }
}

// ---------------------------------------------------------------------------
// Grouped GEMM. Grid: (OUT_FEAT/128, NUM_EXPERT). Block: 256 = 8 wave32s.
// Block = expert e, 128 columns (wave owns 16). Loops over 64-token
// supertiles; A staged in LDS as split bf16 once per (supertile, K-chunk),
// B converted inline from fp32 weights (L2-resident across supertiles).
// ---------------------------------------------------------------------------
__global__ void __launch_bounds__(256)
moe_gemm_kernel(const float* __restrict__ inp, const float* __restrict__ weight,
                const int* __restrict__ ws, float* __restrict__ out) {
  __shared__ __bf16 aHi[M_SUPER][A_STRIDE];
  __shared__ __bf16 aLo[M_SUPER][A_STRIDE];
  __shared__ int    tokS[M_SUPER];

  const int e   = blockIdx.y;
  const int cnt = ws[e];
  if (cnt == 0) return;                 // uniform per block
  const int base = ws[16 + e];
  const int* __restrict__ perm = ws + 64;

  const int tid  = threadIdx.x;
  const int lane = tid & 31;
  const int wave = tid >> 5;
  const int grp  = lane >> 4;           // lane half: selects K sub-ranges
  const int half = lane & 15;
  const int nCol = (blockIdx.x * 8 + wave) * 16 + half;

  // B column nCol == weight row nCol of expert e ([OUT, IN] row-major).
  const float* __restrict__ wrow =
      weight + ((size_t)e * OUT_FEAT + nCol) * IN_FEAT;

  const int sr = tid >> 2;              // staging row   0..63
  const int sk = (tid & 3) * 32;        // staging K off 0/32/64/96

  for (int st = 0; st < cnt; st += M_SUPER) {
    __syncthreads();                    // tokS/LDS reuse across supertiles
    if (tid < M_SUPER) {
      int r = st + tid;
      if (r > cnt - 1) r = cnt - 1;     // clamp partial tile (stores masked)
      tokS[tid] = perm[base + r];
    }
    __syncthreads();

    v8f c0 = {}, c1 = {}, c2 = {}, c3 = {};
    const float* __restrict__ arow = inp + (size_t)tokS[sr] * IN_FEAT + sk;

    for (int kch = 0; kch < IN_FEAT; kch += K_CHUNK) {
      // ---- cooperative stage: A[64 x 128] fp32 -> split bf16 in LDS ----
#pragma unroll
      for (int g = 0; g < 4; ++g) {
        float x[8];
        *(float4*)(&x[0]) = *(const float4*)(arow + kch + g * 8);
        *(float4*)(&x[4]) = *(const float4*)(arow + kch + g * 8 + 4);
        v8bf hv, lv;
#pragma unroll
        for (int i = 0; i < 8; ++i) {
          __bf16 h = (__bf16)x[i];
          hv[i] = h;
          lv[i] = (__bf16)(x[i] - (float)h);
        }
        *(v8bf*)(&aHi[sr][sk + g * 8]) = hv;
        *(v8bf*)(&aLo[sr][sk + g * 8]) = lv;
      }
      __syncthreads();

#pragma unroll
      for (int kc = 0; kc < K_CHUNK; kc += 32) {
        // B 32x16 bf16: lanes 0-15 hold K {kc..kc+7, kc+16..kc+23} of their column,
        // lanes 16-31 shifted by +8 (matches A fragment K order).
        v16bf bhi, blo;
        const float* bp = wrow + kch + kc + grp * 8;
        load_cvt16(bp, bp + 16, bhi, blo);
        __builtin_prefetch(bp + K_CHUNK, 0, 0);  // global_prefetch next chunk

        // A 16x32 bf16 fragment: lanes 0-15 K {kl..kl+7, kl+16..kl+23},
        // lanes 16-31 shifted by +8 (ISA A-matrix layout).
        const int kl = kc + grp * 8;
#pragma unroll
        for (int ms = 0; ms < 4; ++ms) {
          const int row = ms * 16 + half;
          AFrag ah, al;
          ah.h[0] = *(const v8bf*)(&aHi[row][kl]);
          ah.h[1] = *(const v8bf*)(&aHi[row][kl + 16]);
          al.h[0] = *(const v8bf*)(&aLo[row][kl]);
          al.h[1] = *(const v8bf*)(&aLo[row][kl + 16]);
          v8f& c = (ms == 0) ? c0 : (ms == 1) ? c1 : (ms == 2) ? c2 : c3;
          // 3-term bf16 split ~= fp32: AhiBhi + AloBhi + AhiBlo, f32 accum.
          c = __builtin_amdgcn_wmma_f32_16x16x32_bf16(false, ah.v, false, bhi,
                                                      (short)0, c, false, false);
          c = __builtin_amdgcn_wmma_f32_16x16x32_bf16(false, al.v, false, bhi,
                                                      (short)0, c, false, false);
          c = __builtin_amdgcn_wmma_f32_16x16x32_bf16(false, ah.v, false, blo,
                                                      (short)0, c, false, false);
        }
      }
      __syncthreads();                  // before restaging A next K-chunk
    }

    // C/D f32 16x16: VGPR j -> tile row j (lanes 0-15) / j+8 (lanes 16-31).
    const int rem = cnt - st;
#pragma unroll
    for (int ms = 0; ms < 4; ++ms) {
      const v8f& c = (ms == 0) ? c0 : (ms == 1) ? c1 : (ms == 2) ? c2 : c3;
#pragma unroll
      for (int j = 0; j < 8; ++j) {
        int m = ms * 16 + j + grp * 8;
        if (m < rem)
          out[(size_t)tokS[m] * OUT_FEAT + nCol] = c[j];
      }
    }
  }
}

extern "C" void kernel_launch(void* const* d_in, const int* in_sizes, int n_in,
                              void* d_out, int out_size, void* d_ws, size_t ws_size,
                              hipStream_t stream) {
  const float* inp    = (const float*)d_in[0];
  const int*   gate   = (const int*)d_in[1];
  const float* weight = (const float*)d_in[2];
  float*       out    = (float*)d_out;
  int*         ws     = (int*)d_ws;

  moe_setup_kernel<<<1, 256, 0, stream>>>(gate, ws);
  dim3 grid(OUT_FEAT / 128, NUM_EXPERT);
  moe_gemm_kernel<<<grid, 256, 0, stream>>>(inp, weight, ws, out);
}
